// NonlocalInteraction_10995116278397
// MI455X (gfx1250) — matrix operand order repaired and checked by reading, not compiled, not run on hardware
//
#include <hip/hip_runtime.h>
#include <hip/hip_bf16.h>

#define NATOMS 16384
#define FEAT   128
#define KV_TILE 32
#define KS_STRIDE 136   // bf16 elems per K row in LDS (128 + 8 pad) -> 272B, 16B aligned
#define VS_STRIDE 40    // bf16 elems per vT row in LDS (32 + 8 pad) -> 80B, 16B aligned
#define PS_STRIDE 40    // bf16 elems per P row in LDS (32 + 8 pad) -> 80B, 16B aligned

typedef __attribute__((ext_vector_type(16))) __bf16 v16bf;
typedef __attribute__((ext_vector_type(8)))  __bf16 v8bf;
typedef __attribute__((ext_vector_type(8)))  float  v8f;
typedef __attribute__((ext_vector_type(8)))  float  f32x8;
typedef __attribute__((ext_vector_type(4)))  unsigned int u32x4;

static __device__ __forceinline__ v16bf combine16(v8bf lo, v8bf hi) {
  union { v16bf v; v8bf h[2]; } u;
  u.h[0] = lo; u.h[1] = hi;
  return u.v;
}
static __device__ __forceinline__ v8bf cvt8(f32x8 a) {
  v8bf r;
  #pragma unroll
  for (int i = 0; i < 8; ++i) r[i] = (__bf16)a[i];
  return r;
}
static __device__ __forceinline__ v8f zero8() {
  v8f r;
  #pragma unroll
  for (int i = 0; i < 8; ++i) r[i] = 0.f;
  return r;
}
static __device__ __forceinline__ v8f wmma_bf16(v16bf a, v16bf b, v8f c) {
  // D = A(16x32 bf16) * B(32x16 bf16) + C(16x16 f32)
  return __builtin_amdgcn_wmma_f32_16x16x32_bf16(false, a, false, b, (short)0, c,
                                                 false, false);
}

// Async global -> LDS copy, 16B per lane, tracked by ASYNCcnt.
// VDST operand = LDS byte offset; per ISA 10.2 the low 32 bits of the generic
// shared-aperture address ARE the LDS address (high 32 bits are the aperture).
static __device__ __forceinline__ void async_ld16(void* lds_dst, const void* gsrc) {
  unsigned loff = (unsigned)(unsigned long long)lds_dst;
  asm volatile("global_load_async_to_lds_b128 %0, %1, off"
               :: "v"(loff), "v"(gsrc) : "memory");
}
static __device__ __forceinline__ void wait_async0() {
#if __has_builtin(__builtin_amdgcn_s_wait_asynccnt)
  __builtin_amdgcn_s_wait_asynccnt(0);
#else
  asm volatile("s_wait_asynccnt 0x0" ::: "memory");
#endif
}

// ---------------------------------------------------------------------------
// Phase 1: q = (x @ Wq^T) * 1/sqrt(F)  -> bf16 [N,F]
//          k =  x @ Wk^T               -> bf16 [N,F]
//          vT = (x @ Wv^T)^T           -> bf16 [F,N]
// ---------------------------------------------------------------------------
__global__ __launch_bounds__(32, 1)
void qkv_proj_kernel(const float* __restrict__ x,
                     const float* __restrict__ Wq,
                     const float* __restrict__ Wk,
                     const float* __restrict__ Wv,
                     __bf16* __restrict__ q16,
                     __bf16* __restrict__ k16,
                     __bf16* __restrict__ vT16) {
  const int lane = threadIdx.x & 31;
  const int hf   = lane >> 4;
  const int l16  = lane & 15;
  const int n0   = blockIdx.x * 16;
  const int proj = blockIdx.y;             // 0=q, 1=k, 2=v
  const float* W = (proj == 0) ? Wq : ((proj == 1) ? Wk : Wv);
  const float scale = (proj == 0) ? 0.08838834764831845f : 1.0f;

  v16bf a[4];
  const float* xrow = x + (size_t)(n0 + l16) * FEAT;
  #pragma unroll
  for (int kk = 0; kk < 4; ++kk) {
    const int kb = kk * 32;
    f32x8 lo = *(const f32x8*)(xrow + kb + hf * 8);
    f32x8 hi = *(const f32x8*)(xrow + kb + 16 + hf * 8);
    a[kk] = combine16(cvt8(lo), cvt8(hi));
  }

  for (int f0 = 0; f0 < FEAT; f0 += 16) {
    v8f acc = zero8();
    #pragma unroll
    for (int kk = 0; kk < 4; ++kk) {
      const int kb = kk * 32;
      const float* wrow = W + (size_t)(f0 + l16) * FEAT + kb + hf * 8;
      f32x8 blo = *(const f32x8*)(wrow);
      f32x8 bhi = *(const f32x8*)(wrow + 16);
      acc = wmma_bf16(a[kk], combine16(cvt8(blo), cvt8(bhi)), acc);
    }
    #pragma unroll
    for (int r = 0; r < 8; ++r) {
      const __bf16 val = (__bf16)(acc[r] * scale);
      const int mrow = n0 + r + hf * 8;
      if (proj == 0)      q16[(size_t)mrow * FEAT + f0 + l16] = val;
      else if (proj == 1) k16[(size_t)mrow * FEAT + f0 + l16] = val;
      else                vT16[(size_t)(f0 + l16) * NATOMS + mrow] = val;
    }
  }
}

// ---------------------------------------------------------------------------
// Phase 2: flash attention, async-DMA double-buffered K/V^T tiles in LDS.
// Block = 128 threads (4 waves); each wave owns a 16-row Q tile.
// ---------------------------------------------------------------------------
__global__ __launch_bounds__(128, 1)
void flash_attn_kernel(const __bf16* __restrict__ q16,
                       const __bf16* __restrict__ k16,
                       const __bf16* __restrict__ vT16,
                       float* __restrict__ out) {
  __shared__ __bf16 ks[2][KV_TILE * KS_STRIDE];   // 2 x 8704 B
  __shared__ __bf16 vs[2][FEAT * VS_STRIDE];      // 2 x 10240 B
  __shared__ __bf16 ps[4 * 16 * PS_STRIDE];       // 5120 B

  const int tid  = threadIdx.x;
  const int wave = tid >> 5;
  const int lane = tid & 31;
  const int hf   = lane >> 4;
  const int l16  = lane & 15;
  const int q0   = blockIdx.x * 64 + wave * 16;

  // Q tile (16 x 128) in A-operand layout, all 4 k-steps resident.
  v16bf aq[4];
  {
    const __bf16* qrow = q16 + (size_t)(q0 + l16) * FEAT;
    #pragma unroll
    for (int kk = 0; kk < 4; ++kk) {
      const int kb = kk * 32;
      v8bf lo = *(const v8bf*)(qrow + kb + hf * 8);
      v8bf hi = *(const v8bf*)(qrow + kb + 16 + hf * 8);
      aq[kk] = combine16(lo, hi);
    }
  }

  v8f o[8];
  float m[8], lsum[8];
  #pragma unroll
  for (int i = 0; i < 8; ++i) { o[i] = zero8(); m[i] = -__builtin_inff(); lsum[i] = 0.f; }

  __bf16* myp = ps + wave * 16 * PS_STRIDE;

  // issue async loads for the 32-row kv tile starting at `kt` into buffer `b`
  auto issue_tile = [&](int kt, int b) {
    // K tile: 32 rows x 256 B  -> 512 b128 ops -> 4 per thread (16 per row)
    #pragma unroll
    for (int i = 0; i < 4; ++i) {
      const int li  = i * 128 + tid;
      const int row = li >> 4;
      const int cb  = (li & 15) * 16;
      async_ld16((char*)&ks[b][0] + row * (KS_STRIDE * 2) + cb,
                 (const char*)k16 + (size_t)(kt + row) * (FEAT * 2) + cb);
    }
    // vT tile: 128 rows x 64 B -> 512 b128 ops -> 4 per thread (4 per row)
    #pragma unroll
    for (int i = 0; i < 4; ++i) {
      const int li  = i * 128 + tid;
      const int row = li >> 2;
      const int cb  = (li & 3) * 16;
      async_ld16((char*)&vs[b][0] + row * (VS_STRIDE * 2) + cb,
                 (const char*)vT16 + ((size_t)row * NATOMS + kt) * 2 + cb);
    }
  };

  issue_tile(0, 0);
  int buf = 0;

  for (int kt = 0; kt < NATOMS; kt += KV_TILE, buf ^= 1) {
    wait_async0();       // my async ops for tile kt have landed in LDS
    __syncthreads();     // everyone's landed; everyone done reading other buffer
    if (kt + KV_TILE < NATOMS) issue_tile(kt + KV_TILE, buf ^ 1);

    const __bf16* ksb = &ks[buf][0];
    const __bf16* vsb = &vs[buf][0];

    // ---- S = Q * K^T : 32 kv columns as two 16x16 C tiles
    v8f c0 = zero8(), c1 = zero8();
    #pragma unroll
    for (int kk = 0; kk < 4; ++kk) {
      const int kb = kk * 32;
      const __bf16* kp0 = ksb + l16 * KS_STRIDE + kb + hf * 8;
      c0 = wmma_bf16(aq[kk],
                     combine16(*(const v8bf*)kp0, *(const v8bf*)(kp0 + 16)), c0);
      const __bf16* kp1 = ksb + (16 + l16) * KS_STRIDE + kb + hf * 8;
      c1 = wmma_bf16(aq[kk],
                     combine16(*(const v8bf*)kp1, *(const v8bf*)(kp1 + 16)), c1);
    }

    // ---- online softmax over these 32 columns
    float alpha[8];
    #pragma unroll
    for (int r = 0; r < 8; ++r) {
      float t = fmaxf(c0[r], c1[r]);
      #pragma unroll
      for (int msk = 1; msk < 16; msk <<= 1)
        t = fmaxf(t, __shfl_xor(t, msk, 32));
      const float mn = fmaxf(m[r], t);
      alpha[r] = __expf(m[r] - mn);
      m[r] = mn;
      const float p0 = __expf(c0[r] - mn);
      const float p1 = __expf(c1[r] - mn);
      float rs = p0 + p1;
      #pragma unroll
      for (int msk = 1; msk < 16; msk <<= 1)
        rs += __shfl_xor(rs, msk, 32);
      lsum[r] = lsum[r] * alpha[r] + rs;
      const int prow = r + hf * 8;
      myp[prow * PS_STRIDE + l16]      = (__bf16)p0;
      myp[prow * PS_STRIDE + 16 + l16] = (__bf16)p1;
    }

    // rescale accumulator
    #pragma unroll
    for (int c = 0; c < 8; ++c)
      #pragma unroll
      for (int r = 0; r < 8; ++r)
        o[c][r] *= alpha[r];

    // ---- reload P in A-operand layout (same-wave LDS ops stay in order)
    const __bf16* prw = myp + l16 * PS_STRIDE;
    v16bf pa = combine16(*(const v8bf*)(prw + hf * 8),
                         *(const v8bf*)(prw + 16 + hf * 8));

    // ---- O += P * V  (8 feature chunks of 16)
    #pragma unroll
    for (int c = 0; c < 8; ++c) {
      const __bf16* vp = vsb + (c * 16 + l16) * VS_STRIDE + hf * 8;
      o[c] = wmma_bf16(pa, combine16(*(const v8bf*)vp, *(const v8bf*)(vp + 16)),
                       o[c]);
    }
  }

  // epilogue: out = O / l
  const int mbase = q0 + hf * 8;
  #pragma unroll
  for (int r = 0; r < 8; ++r) {
    const float inv = 1.0f / lsum[r];
    float* orow = out + (size_t)(mbase + r) * FEAT + l16;
    #pragma unroll
    for (int c = 0; c < 8; ++c)
      orow[c * 16] = o[c][r] * inv;
  }
}

// ---------------------------------------------------------------------------
extern "C" void kernel_launch(void* const* d_in, const int* in_sizes, int n_in,
                              void* d_out, int out_size, void* d_ws, size_t ws_size,
                              hipStream_t stream) {
  const float* x  = (const float*)d_in[0];
  const float* Wq = (const float*)d_in[1];
  const float* Wk = (const float*)d_in[2];
  const float* Wv = (const float*)d_in[3];
  float* out = (float*)d_out;

  __bf16* q16  = (__bf16*)d_ws;                         // 4 MB
  __bf16* k16  = q16 + (size_t)NATOMS * FEAT;           // 4 MB
  __bf16* vT16 = k16 + (size_t)NATOMS * FEAT;           // 4 MB (transposed [F,N])

  dim3 gProj(NATOMS / 16, 3);
  qkv_proj_kernel<<<gProj, 32, 0, stream>>>(x, Wq, Wk, Wv, q16, k16, vT16);
  flash_attn_kernel<<<NATOMS / KV_TILE, 128, 0, stream>>>(q16, k16, vT16, out);
}
